// CompatSelfAttention_41360535061187
// MI455X (gfx1250) — matrix operand order repaired and checked by reading, not compiled, not run on hardware
//
#include <hip/hip_runtime.h>

// ---------------------------------------------------------------------------
// CDNA5 (gfx1250) wave32 WMMA self-attention:
//   qkv = x @ Wqkv (bf16 WMMA, f32 accum)  -> ws
//   y   = flash-attention(q,k,v)           -> ws  (online softmax, WMMA, TDM)
//   out = y @ Wproj                        -> d_out (f32)
// ---------------------------------------------------------------------------

typedef __attribute__((ext_vector_type(16))) __bf16         vbf16;
typedef __attribute__((ext_vector_type(16))) unsigned short v16u;
typedef __attribute__((ext_vector_type(8)))  float          v8f;
typedef __attribute__((ext_vector_type(4)))  unsigned int   u32x4;
typedef __attribute__((ext_vector_type(8)))  int            i32x8;
typedef __attribute__((ext_vector_type(4)))  int            i32x4;

#if defined(__has_builtin)
#if __has_builtin(__builtin_amdgcn_tensor_load_to_lds) && \
    __has_builtin(__builtin_amdgcn_s_wait_tensorcnt)
#define USE_TDM 1
#endif
#endif
#ifndef USE_TDM
#define USE_TDM 0
#endif

__device__ __forceinline__ unsigned short f2bf(float f) {
  unsigned int u = __float_as_uint(f);
  u += 0x7FFFu + ((u >> 16) & 1u);  // round to nearest even
  return (unsigned short)(u >> 16);
}

// load 16 contiguous bf16 (32B) as one fragment
__device__ __forceinline__ v16u ld32B(const unsigned short* p) {
  union { uint4 q[2]; v16u v; } t;
  t.q[0] = *(const uint4*)(p);
  t.q[1] = *(const uint4*)(p + 8);
  return t.v;
}

__device__ __forceinline__ v16u ld2x16B(const unsigned short* p) {
  union { uint4 q[2]; v16u v; } t;
  t.q[0] = *(const uint4*)(p);
  t.q[1] = *(const uint4*)(p + 16);
  return t.v;
}

__device__ __forceinline__ v8f wmma_bf16(v16u a, v16u b, v8f c) {
  return __builtin_amdgcn_wmma_f32_16x16x32_bf16(
      false, __builtin_bit_cast(vbf16, a),
      false, __builtin_bit_cast(vbf16, b),
      (short)0, c, false, false);
}

// ---------------------------------------------------------------------------
// GEMM: D[M,N] = A[M,K] * B[K,N].  A: f32 or bf16; B: f32; D: bf16 or f32.
// 256 threads (8 waves), block tile 64x128, wave tile 16x64 (4 WMMA / K-step).
// ---------------------------------------------------------------------------
template <bool A_BF16, bool OUT_BF16>
__global__ __launch_bounds__(256) void gemm_bf16_wmma(
    const void* __restrict__ Ap, const float* __restrict__ Bp,
    void* __restrict__ Dp, int M, int N, int K) {
  __shared__ unsigned short lA[64 * 32];    // [m][k]
  __shared__ unsigned short lBt[128 * 32];  // [n][k]

  const int tid  = threadIdx.x;
  const int lane = tid & 31;
  const int wave = tid >> 5;
  const int h    = lane >> 4;
  const int l16  = lane & 15;

  const int m0 = blockIdx.y * 64;
  const int n0 = blockIdx.x * 128;
  const int wm = (wave & 3) * 16;   // wave M sub-tile
  const int wn = (wave >> 2) * 64;  // wave N sub-tile (4 x 16)

  v8f acc[4];
  {
    v8f z = {};
#pragma unroll
    for (int t = 0; t < 4; ++t) acc[t] = z;
  }

  for (int k0 = 0; k0 < K; k0 += 32) {
    // ---- stage A 64x32 ----
    for (int i = tid; i < 512; i += 256) {
      const int row = i >> 3;
      const int col = (i & 7) * 4;
      if (A_BF16) {
        const unsigned short* src =
            (const unsigned short*)Ap + (size_t)(m0 + row) * K + k0 + col;
        *(ushort4*)&lA[row * 32 + col] = *(const ushort4*)src;
      } else {
        const float* src = (const float*)Ap + (size_t)(m0 + row) * K + k0 + col;
        float4 f = *(const float4*)src;
        *(ushort4*)&lA[row * 32 + col] =
            make_ushort4(f2bf(f.x), f2bf(f.y), f2bf(f.z), f2bf(f.w));
      }
    }
    // ---- stage B^T 128x32: lBt[n][k] = B[k0+k][n0+n] ----
    for (int i = tid; i < 1024; i += 256) {
      const int kk = i >> 5;
      const int nn = (i & 31) * 4;
      const float* src = Bp + (size_t)(k0 + kk) * N + n0 + nn;
      float4 f = *(const float4*)src;
      lBt[(nn + 0) * 32 + kk] = f2bf(f.x);
      lBt[(nn + 1) * 32 + kk] = f2bf(f.y);
      lBt[(nn + 2) * 32 + kk] = f2bf(f.z);
      lBt[(nn + 3) * 32 + kk] = f2bf(f.w);
    }
    // ---- prefetch next K-tile into L2/WGP$ (global_prefetch_b8) ----
    if (k0 + 32 < K) {
      const int row = tid >> 3, col = (tid & 7) * 4;
      if (A_BF16)
        __builtin_prefetch(
            (const unsigned short*)Ap + (size_t)(m0 + row) * K + k0 + 32 + col,
            0, 2);
      else
        __builtin_prefetch(
            (const float*)Ap + (size_t)(m0 + row) * K + k0 + 32 + col, 0, 2);
      __builtin_prefetch(
          Bp + (size_t)(k0 + 32 + (tid >> 5)) * N + n0 + (tid & 31) * 4, 0, 2);
    }
    __syncthreads();

    // A fragment: row = wm + l16; elems 0..7 -> K 8h.., 8..15 -> K 16+8h..
    v16u af;
    {
      union { uint4 q[2]; v16u v; } t;
      t.q[0] = *(const uint4*)&lA[(wm + l16) * 32 + 8 * h];
      t.q[1] = *(const uint4*)&lA[(wm + l16) * 32 + 16 + 8 * h];
      af = t.v;
    }
#pragma unroll
    for (int t = 0; t < 4; ++t) {
      v16u bf = ld2x16B(&lBt[(wn + 16 * t + l16) * 32 + 8 * h]);
      acc[t] = wmma_bf16(af, bf, acc[t]);
    }
    __syncthreads();
  }

  // C/D layout: VGPR r, lanes 0-15 -> M=r, lanes 16-31 -> M=r+8; N = l16
#pragma unroll
  for (int t = 0; t < 4; ++t) {
#pragma unroll
    for (int r = 0; r < 8; ++r) {
      const int row = m0 + wm + r + 8 * h;
      const int col = n0 + wn + 16 * t + l16;
      if (OUT_BF16)
        ((unsigned short*)Dp)[(size_t)row * N + col] = f2bf(acc[t][r]);
      else
        ((float*)Dp)[(size_t)row * N + col] = acc[t][r];
    }
  }
}

// ---------------------------------------------------------------------------
// Flash attention: 256 threads = 8 waves, 16 queries/wave, 128 queries/block.
// Key blocks of 32; K staged via TDM [key][e], V staged transposed [e][key].
// Per-wave O accumulator: 16x512 f32 (32 x v8f).
// ---------------------------------------------------------------------------
__global__ __launch_bounds__(256) void attn_wmma(
    const unsigned short* __restrict__ qkv, unsigned short* __restrict__ y) {
  constexpr int T = 4096, C = 512, C3 = 1536;
  __shared__ unsigned short lKV[32 * 512];   // 32 KB; K then (re-used) V^T
  __shared__ unsigned short lP[8][16 * 32];  // per-wave P layout shuffle

  const int tid  = threadIdx.x;
  const int lane = tid & 31;
  const int wave = tid >> 5;
  const int h    = lane >> 4;
  const int l16  = lane & 15;

  const int wg   = blockIdx.x;
  const int b    = wg >> 5;
  const int qblk = wg & 31;
  const int q0   = qblk * 128;
  const int qrow = q0 + wave * 16;
  const size_t base = (size_t)b * T * C3;

  // ---- preload q as 16 A-fragments ----
  v16u qf[16];
#pragma unroll
  for (int kc = 0; kc < 16; ++kc) {
    const unsigned short* p =
        qkv + base + (size_t)(qrow + l16) * C3 + kc * 32 + 8 * h;
    union { uint4 q[2]; v16u v; } t;
    t.q[0] = *(const uint4*)p;
    t.q[1] = *(const uint4*)(p + 16);
    qf[kc] = t.v;
  }

  v8f acc[32];
  {
    v8f z = {};
#pragma unroll
    for (int n = 0; n < 32; ++n) acc[n] = z;
  }
  float m[8], l[8];
#pragma unroll
  for (int r = 0; r < 8; ++r) { m[r] = -1e30f; l[r] = 0.0f; }

  const float scale = 0.044194173824159216f;  // 1/sqrt(512)
  const int nkb = qblk * 4 + 4;

  for (int kb = 0; kb < nkb; ++kb) {
    const int s0 = kb * 32;

    // ---- prefetch next key block (K and V rows) ----
    if (kb + 1 < nkb) {
      const unsigned short* nk =
          qkv + base + (size_t)(s0 + 32 + (tid >> 3)) * C3 + 512 +
          (tid & 7) * 64;
      __builtin_prefetch(nk, 0, 2);        // next K chunk
      __builtin_prefetch(nk + 512, 0, 2);  // next V chunk
    }

    // ---- stage K block [key][e] ----
#if USE_TDM
    if (wave == 0) {
      const unsigned long long ga =
          (unsigned long long)(uintptr_t)(qkv + base + (size_t)s0 * C3 + 512);
      union { unsigned int w[4]; u32x4 v; } g0 = {};
      g0.w[0] = 1u;  // count=1, user descriptor
      g0.w[1] = (unsigned int)(uintptr_t)(&lKV[0]);        // lds_addr
      g0.w[2] = (unsigned int)ga;                          // global_addr lo
      g0.w[3] = (unsigned int)((ga >> 32) & 0x1FFFFFFu) |  // global_addr hi
                0x80000000u;                               // type=2 (image)
      union { int w[8]; i32x8 v; } g1 = {};
      g1.w[0] = 1 << 16;                       // data_size = 2 bytes
      g1.w[1] = (int)(((unsigned)C & 0xFFFFu) << 16);          // tensor_dim0 lo
      g1.w[2] = (int)((((unsigned)C >> 16) & 0xFFFFu) |        // tensor_dim0 hi
                      (((unsigned)T & 0xFFFFu) << 16));        // tensor_dim1 lo
      g1.w[3] = (int)((((unsigned)T >> 16) & 0xFFFFu) |        // tensor_dim1 hi
                      ((unsigned)C << 16));                    // tile_dim0 = 512
      g1.w[4] = 32;                            // tile_dim1 = 32, tile_dim2 = 0
      g1.w[5] = C3;                            // tensor_dim0_stride = 1536
      g1.w[6] = 0;
      g1.w[7] = 0;
      i32x4 z4 = {};
      i32x8 z8 = {};
      __builtin_amdgcn_tensor_load_to_lds(g0.v, g1.v, z4, z4, z8, 0);
      __builtin_amdgcn_s_wait_tensorcnt(0);
    }
#else
    for (int i = tid; i < 4096; i += 256) {
      const int key = i >> 7;
      const int e   = (i & 127) * 4;
      *(ushort4*)&lKV[key * 512 + e] =
          *(const ushort4*)(qkv + base + (size_t)(s0 + key) * C3 + 512 + e);
    }
#endif
    __syncthreads();

    // ---- S = q @ K^T (two 16x16 tiles over 32 keys) ----
    v8f sA = {}, sB = {};
#pragma unroll
    for (int kc = 0; kc < 16; ++kc) {
      v16u bk0 = ld2x16B(&lKV[l16 * 512 + kc * 32 + 8 * h]);
      v16u bk1 = ld2x16B(&lKV[(16 + l16) * 512 + kc * 32 + 8 * h]);
      sA = wmma_bf16(qf[kc], bk0, sA);
      sB = wmma_bf16(qf[kc], bk1, sB);
    }

    // ---- causal mask + online softmax update ----
    float p0[8], p1[8], alpha[8];
#pragma unroll
    for (int r = 0; r < 8; ++r) {
      const int qi = qrow + r + 8 * h;
      float sa = (s0 + l16 <= qi)      ? sA[r] * scale : -1e30f;
      float sb = (s0 + 16 + l16 <= qi) ? sB[r] * scale : -1e30f;
      float mx = fmaxf(sa, sb);
      mx = fmaxf(mx, __shfl_xor(mx, 1));
      mx = fmaxf(mx, __shfl_xor(mx, 2));
      mx = fmaxf(mx, __shfl_xor(mx, 4));
      mx = fmaxf(mx, __shfl_xor(mx, 8));
      const float mn = fmaxf(m[r], mx);
      const float al = __expf(m[r] - mn);
      const float pa = __expf(sa - mn);
      const float pb = __expf(sb - mn);
      float rs = pa + pb;
      rs += __shfl_xor(rs, 1);
      rs += __shfl_xor(rs, 2);
      rs += __shfl_xor(rs, 4);
      rs += __shfl_xor(rs, 8);
      l[r] = l[r] * al + rs;
      m[r] = mn;
      alpha[r] = al;
      p0[r] = pa;
      p1[r] = pb;
    }
#pragma unroll
    for (int n = 0; n < 32; ++n) {
#pragma unroll
      for (int r = 0; r < 8; ++r) acc[n][r] *= alpha[r];
    }

    // ---- P: C/D layout -> A layout via per-wave LDS scratch ----
#pragma unroll
    for (int r = 0; r < 8; ++r) {
      lP[wave][(r + 8 * h) * 32 + l16]      = f2bf(p0[r]);
      lP[wave][(r + 8 * h) * 32 + 16 + l16] = f2bf(p1[r]);
    }
    __builtin_amdgcn_wave_barrier();
    v16u pf;
    {
      union { uint4 q[2]; v16u v; } t;
      t.q[0] = *(const uint4*)&lP[wave][l16 * 32 + 8 * h];
      t.q[1] = *(const uint4*)&lP[wave][l16 * 32 + 16 + 8 * h];
      pf = t.v;
    }
    __syncthreads();

    // ---- stage V transposed [e][key] into the same buffer ----
    for (int i = tid; i < 4096; i += 256) {
      const int key = i >> 7;
      const int e   = (i & 127) * 4;
      ushort4 v4 =
          *(const ushort4*)(qkv + base + (size_t)(s0 + key) * C3 + 1024 + e);
      lKV[(e + 0) * 32 + key] = v4.x;
      lKV[(e + 1) * 32 + key] = v4.y;
      lKV[(e + 2) * 32 + key] = v4.z;
      lKV[(e + 3) * 32 + key] = v4.w;
    }
    __syncthreads();

    // ---- O += P @ V (32 WMMAs: one per 16-wide embed tile) ----
#pragma unroll
    for (int n = 0; n < 32; ++n) {
      v16u bv = ld2x16B(&lKV[(n * 16 + l16) * 32 + 8 * h]);
      acc[n] = wmma_bf16(pf, bv, acc[n]);
    }
    __syncthreads();
  }

  // ---- epilogue: y = O / l, bf16 ----
  const size_t ybase = (size_t)b * T * C;
#pragma unroll
  for (int n = 0; n < 32; ++n) {
#pragma unroll
    for (int r = 0; r < 8; ++r) {
      const int row = qrow + r + 8 * h;
      y[ybase + (size_t)row * C + n * 16 + l16] = f2bf(acc[n][r] / l[r]);
    }
  }
}

// ---------------------------------------------------------------------------
extern "C" void kernel_launch(void* const* d_in, const int* in_sizes, int n_in,
                              void* d_out, int out_size, void* d_ws,
                              size_t ws_size, hipStream_t stream) {
  (void)in_sizes; (void)n_in; (void)out_size; (void)ws_size;
  constexpr int B = 4, T = 4096, C = 512, C3 = 1536;
  constexpr int M = B * T;  // 16384

  const float* x     = (const float*)d_in[0];
  const float* Wqkv  = (const float*)d_in[1];
  const float* Wproj = (const float*)d_in[2];
  float* out = (float*)d_out;

  unsigned short* qkv = (unsigned short*)d_ws;   // [M, 1536] bf16
  unsigned short* y   = qkv + (size_t)M * C3;    // [M,  512] bf16

  dim3 blk(256);
  gemm_bf16_wmma<false, true>
      <<<dim3(C3 / 128, M / 64), blk, 0, stream>>>(x, Wqkv, qkv, M, C3, C);
  attn_wmma<<<dim3(128), blk, 0, stream>>>(qkv, y);
  gemm_bf16_wmma<true, false>
      <<<dim3(C / 128, M / 64), blk, 0, stream>>>(y, Wproj, out, M, C, C);
}
